// RelativeMultiHeadAttention_50749333570140
// MI455X (gfx1250) — compile-verified, matrix operand order checked by reading, not run on hardware
//
#include <hip/hip_runtime.h>
#include <hip/hip_bf16.h>

// ---------------- types ----------------
typedef __bf16 bf16_t;
typedef __attribute__((ext_vector_type(16))) bf16_t v16bf;
typedef __attribute__((ext_vector_type(8)))  bf16_t v8bf;
typedef __attribute__((ext_vector_type(8)))  float  v8f;

// ---------------- problem constants ----------------
constexpr int D      = 2048;   // d_model
constexpr int BATCH  = 2;
constexpr int SEQ    = 4096;
constexpr int M_ROWS = BATCH * SEQ;   // 8192
constexpr int N_QKV  = 3 * D;         // 6144

// workspace layout (bytes)
constexpr size_t OFF_QKVF = 0;                                       // fp32 [8192 x 6144]
constexpr size_t OFF_XB   = OFF_QKVF + (size_t)M_ROWS * N_QKV * 4;   // bf16 [8192 x 2048]
constexpr size_t OFF_W1T  = OFF_XB   + (size_t)M_ROWS * D * 2;       // bf16 [6144 x 2048]
constexpr size_t OFF_W2T  = OFF_W1T  + (size_t)N_QKV * D * 2;        // bf16 [2048 x 2048]
constexpr size_t OFF_CTXB = OFF_W2T  + (size_t)D * D * 2;            // bf16 [8192 x 2048]

// ---------------- CDNA5 async global->LDS support ----------------
#if __has_builtin(__builtin_amdgcn_global_load_async_to_lds_b128)
#define ASYNC_COPY 1
#define AS_GLOBAL __attribute__((address_space(1)))
#define AS_LOCAL  __attribute__((address_space(3)))
typedef int v4i_vs __attribute__((vector_size(4 * sizeof(int))));
__device__ __forceinline__ void async_copy_b128(const bf16_t* g, bf16_t* l) {
    __builtin_amdgcn_global_load_async_to_lds_b128(
        (AS_GLOBAL v4i_vs*)(g), (AS_LOCAL v4i_vs*)(l), 0, 0);
}
__device__ __forceinline__ void wait_async0() {
#if __has_builtin(__builtin_amdgcn_s_wait_asynccnt)
    __builtin_amdgcn_s_wait_asynccnt(0);
#else
    asm volatile("s_wait_asynccnt 0x0" ::: "memory");
#endif
}
#endif

// ---------------- fp32 -> bf16 convert (row-major copy) ----------------
__global__ void cvt_f32_bf16(const float* __restrict__ in, bf16_t* __restrict__ out, int n8) {
    for (int i = blockIdx.x * blockDim.x + threadIdx.x; i < n8; i += gridDim.x * blockDim.x) {
        const float4* p = reinterpret_cast<const float4*>(in) + (size_t)i * 2;
        float4 x0 = p[0], x1 = p[1];
        v8bf o;
        o[0] = (bf16_t)x0.x; o[1] = (bf16_t)x0.y; o[2] = (bf16_t)x0.z; o[3] = (bf16_t)x0.w;
        o[4] = (bf16_t)x1.x; o[5] = (bf16_t)x1.y; o[6] = (bf16_t)x1.z; o[7] = (bf16_t)x1.w;
        reinterpret_cast<v8bf*>(out)[i] = o;
    }
}

// ---------------- fp32 [K x N] -> bf16 [N x K] transpose ----------------
__global__ void transpose_f32_bf16(const float* __restrict__ in, bf16_t* __restrict__ out,
                                   int K, int N) {
    __shared__ float tile[32][33];
    int bx = blockIdx.x * 32;           // n base
    int by = blockIdx.y * 32;           // k base
    int tx = threadIdx.x, ty = threadIdx.y;   // 32 x 8
#pragma unroll
    for (int j = 0; j < 4; ++j)
        tile[ty + j * 8][tx] = in[(size_t)(by + ty + j * 8) * N + bx + tx];
    __syncthreads();
#pragma unroll
    for (int j = 0; j < 4; ++j)
        out[(size_t)(bx + ty + j * 8) * K + by + tx] = (bf16_t)tile[tx][ty + j * 8];
}

// ---------------- bf16 WMMA GEMM:  C[M,N] = A[M,K] * Bt[N,K]^T + bias[N] ----------------
// block tile 128x128, BK=32, 256 threads = 8 waves (2 x 4), wave tile 64x32.
// Double-buffered LDS fed by CDNA5 async global->LDS copies (ASYNCcnt), with a
// software-pipelined register-staging fallback when the builtin is unavailable.
__device__ __forceinline__ v16bf load_frag(const bf16_t* base) {
    v8bf lo = *reinterpret_cast<const v8bf*>(base);        // K 0..7   (or 8..15 for upper half-wave)
    v8bf hi = *reinterpret_cast<const v8bf*>(base + 16);   // K 16..23 (or 24..31)
    v16bf r;
#pragma unroll
    for (int e = 0; e < 8; ++e) { r[e] = lo[e]; r[8 + e] = hi[e]; }
    return r;
}

__global__ __launch_bounds__(256) void gemm_bf16_wmma(
        const bf16_t* __restrict__ A,   // [Mdim x Kdim] row-major
        const bf16_t* __restrict__ Bt,  // [Ndim x Kdim] row-major (B transposed)
        const float*  __restrict__ bias,
        float* __restrict__ C,          // [Mdim x Ndim]
        int Mdim, int Ndim, int Kdim) {
    __shared__ alignas(16) bf16_t As[2][128 * 32];
    __shared__ alignas(16) bf16_t Bs[2][128 * 32];

    const int t      = threadIdx.x;
    const int waveId = t >> 5;
    const int lane   = t & 31;
    const int r16    = lane & 15;
    const int half   = lane >> 4;
    const int waveM  = waveId & 1;   // 0..1
    const int waveN  = waveId >> 1;  // 0..3
    const int blockM = blockIdx.y * 128;
    const int blockN = blockIdx.x * 128;

    // per-thread staging assignment: 2 x 16B chunks of A and of B per tile
    int row0 = t >> 2;            // chunk t
    int c80  = (t & 3) * 8;
    int row1 = (t + 256) >> 2;    // chunk t+256
    int c81  = ((t + 256) & 3) * 8;

    v8f acc[4][2] = {};
    const int nK = Kdim / 32;

#if ASYNC_COPY
    // ---- async double-buffered pipeline ----
    {   // prefetch tile 0 into buffer 0
        async_copy_b128(A + (size_t)(blockM + row0) * Kdim + c80, &As[0][row0 * 32 + c80]);
        async_copy_b128(A + (size_t)(blockM + row1) * Kdim + c81, &As[0][row1 * 32 + c81]);
        async_copy_b128(Bt + (size_t)(blockN + row0) * Kdim + c80, &Bs[0][row0 * 32 + c80]);
        async_copy_b128(Bt + (size_t)(blockN + row1) * Kdim + c81, &Bs[0][row1 * 32 + c81]);
    }
    for (int kt = 0; kt < nK; ++kt) {
        const int cur = kt & 1;
        wait_async0();          // my writes into buffer `cur` have landed
        __syncthreads();        // everyone's writes landed; everyone done reading `cur^1`
        if (kt + 1 < nK) {      // prefetch next tile into the other buffer
            const int kb = (kt + 1) * 32;
            async_copy_b128(A + (size_t)(blockM + row0) * Kdim + kb + c80, &As[cur ^ 1][row0 * 32 + c80]);
            async_copy_b128(A + (size_t)(blockM + row1) * Kdim + kb + c81, &As[cur ^ 1][row1 * 32 + c81]);
            async_copy_b128(Bt + (size_t)(blockN + row0) * Kdim + kb + c80, &Bs[cur ^ 1][row0 * 32 + c80]);
            async_copy_b128(Bt + (size_t)(blockN + row1) * Kdim + kb + c81, &Bs[cur ^ 1][row1 * 32 + c81]);
        }
        const bf16_t* Abuf = As[cur];
        const bf16_t* Bbuf = Bs[cur];
#else
    // ---- register-staged software pipeline (fallback) ----
    uint4 ra0, ra1, rb0, rb1;
    ra0 = *reinterpret_cast<const uint4*>(A + (size_t)(blockM + row0) * Kdim + c80);
    ra1 = *reinterpret_cast<const uint4*>(A + (size_t)(blockM + row1) * Kdim + c81);
    rb0 = *reinterpret_cast<const uint4*>(Bt + (size_t)(blockN + row0) * Kdim + c80);
    rb1 = *reinterpret_cast<const uint4*>(Bt + (size_t)(blockN + row1) * Kdim + c81);
    for (int kt = 0; kt < nK; ++kt) {
        __syncthreads();        // previous compute done; safe to overwrite LDS
        *reinterpret_cast<uint4*>(&As[0][row0 * 32 + c80]) = ra0;
        *reinterpret_cast<uint4*>(&As[0][row1 * 32 + c81]) = ra1;
        *reinterpret_cast<uint4*>(&Bs[0][row0 * 32 + c80]) = rb0;
        *reinterpret_cast<uint4*>(&Bs[0][row1 * 32 + c81]) = rb1;
        if (kt + 1 < nK) {      // issue next tile's global loads; they fly during compute
            const int kb = (kt + 1) * 32;
            ra0 = *reinterpret_cast<const uint4*>(A + (size_t)(blockM + row0) * Kdim + kb + c80);
            ra1 = *reinterpret_cast<const uint4*>(A + (size_t)(blockM + row1) * Kdim + kb + c81);
            rb0 = *reinterpret_cast<const uint4*>(Bt + (size_t)(blockN + row0) * Kdim + kb + c80);
            rb1 = *reinterpret_cast<const uint4*>(Bt + (size_t)(blockN + row1) * Kdim + kb + c81);
        }
        __syncthreads();
        const bf16_t* Abuf = As[0];
        const bf16_t* Bbuf = Bs[0];
#endif
        // ---- 8 WMMAs on the staged tile ----
        v16bf bfrag[2];
#pragma unroll
        for (int j = 0; j < 2; ++j) {
            int nLocal = waveN * 32 + j * 16 + r16;
            bfrag[j] = load_frag(&Bbuf[nLocal * 32 + half * 8]);
        }
#pragma unroll
        for (int i = 0; i < 4; ++i) {
            int mLocal = waveM * 64 + i * 16 + r16;
            v16bf afrag = load_frag(&Abuf[mLocal * 32 + half * 8]);
#pragma unroll
            for (int j = 0; j < 2; ++j) {
                acc[i][j] = __builtin_amdgcn_wmma_f32_16x16x32_bf16(
                    false, afrag, false, bfrag[j], (short)0, acc[i][j], false, false);
            }
        }
    }

    // epilogue: bias + store (C/D layout: vgpr v -> M = v + half*8; lane%16 -> N)
#pragma unroll
    for (int j = 0; j < 2; ++j) {
        int nG = blockN + waveN * 32 + j * 16 + r16;
        float bv = bias[nG];
#pragma unroll
        for (int i = 0; i < 4; ++i) {
            int mBase = blockM + waveM * 64 + i * 16 + half * 8;
            v8f c = acc[i][j];
#pragma unroll
            for (int v = 0; v < 8; ++v)
                C[(size_t)(mBase + v) * Ndim + nG] = c[v] + bv;
        }
    }
}

// ---------------- per-token 16x16 attention + permuted bf16 store ----------------
// qkv row layout per head h: [q(128) | k(128) | v(128)] at offset h*384.
// Output permutation (faithful swapaxes+reshape): ctx[b,h,s,:] -> row b*4096 + h*256 + s/16,
// columns (s%16)*128 .. +127.
__global__ __launch_bounds__(256) void attn_kernel(const float* __restrict__ qkv,
                                                   const int* __restrict__ mask,
                                                   bf16_t* __restrict__ ctxb) {
    __shared__ float sQKV[N_QKV];       // 24 KB: this token's q,k,v
    __shared__ float sS[256];           // scores / probs
    __shared__ float sMax[16], sInv[16];

    const int tok = blockIdx.x;
    const int b   = tok >> 12;
    const int s   = tok & 4095;
    const int tid = threadIdx.x;

    {   // stage qkv row into LDS (1536 float4's by 256 threads)
        const float4* src = reinterpret_cast<const float4*>(qkv + (size_t)tok * N_QKV);
        float4* dst = reinterpret_cast<float4*>(sQKV);
        for (int i = tid; i < N_QKV / 4; i += 256) dst[i] = src[i];
    }
    __syncthreads();

    // scores[h][g] = (q_h . k_g) / sqrt(D), masked
    {
        const int h = tid >> 4, g = tid & 15;
        const float* qp = &sQKV[h * 384];
        const float* kp = &sQKV[g * 384 + 128];
        float accv = 0.f;
#pragma unroll 8
        for (int d = 0; d < 128; ++d) accv += qp[d] * kp[d];
        accv *= 0.022097086912079608f;          // 1/sqrt(2048)
        if (mask[b * 256 + h * 16 + g] == 0) accv = -__builtin_inff();
        sS[tid] = accv;
    }
    __syncthreads();

    if (tid < 16) {                     // row max + sum(exp)
        float m = sS[tid * 16];
#pragma unroll
        for (int g = 1; g < 16; ++g) m = fmaxf(m, sS[tid * 16 + g]);
        float sum = 0.f;
#pragma unroll
        for (int g = 0; g < 16; ++g) sum += __expf(sS[tid * 16 + g] - m);
        sMax[tid] = m;
        sInv[tid] = 1.f / sum;
    }
    __syncthreads();

    {   // normalize probabilities in place
        const int h = tid >> 4;
        sS[tid] = __expf(sS[tid] - sMax[h]) * sInv[h];
    }
    __syncthreads();

    // ctx[h][:] = attn[h] @ V ; each thread emits 8 contiguous outputs
    {
        const int h = tid >> 4;
        const int dbase = (tid & 15) * 8;
        float o[8] = {0.f, 0.f, 0.f, 0.f, 0.f, 0.f, 0.f, 0.f};
#pragma unroll
        for (int g = 0; g < 16; ++g) {
            float p = sS[h * 16 + g];
            const float* vp = &sQKV[g * 384 + 256 + dbase];
#pragma unroll
            for (int d = 0; d < 8; ++d) o[d] += p * vp[d];
        }
        const int row = b * 4096 + h * 256 + (s >> 4);
        const int col = (s & 15) * 128 + dbase;
        v8bf ov;
#pragma unroll
        for (int d = 0; d < 8; ++d) ov[d] = (bf16_t)o[d];
        *reinterpret_cast<v8bf*>(ctxb + (size_t)row * D + col) = ov;
    }
}

// ---------------- launch ----------------
extern "C" void kernel_launch(void* const* d_in, const int* in_sizes, int n_in,
                              void* d_out, int out_size, void* d_ws, size_t ws_size,
                              hipStream_t stream) {
    const float* query  = (const float*)d_in[0];
    // d_in[1] key, d_in[2] value, d_in[3] pos_embedding: unused (faithful to reference)
    const int*   mask   = (const int*)d_in[4];
    const float* qkv_w  = (const float*)d_in[5];
    const float* qkv_b  = (const float*)d_in[6];
    const float* out_w  = (const float*)d_in[7];
    const float* out_b  = (const float*)d_in[8];

    char* ws = (char*)d_ws;
    float*  qkvf = (float*)(ws + OFF_QKVF);
    bf16_t* Xb   = (bf16_t*)(ws + OFF_XB);
    bf16_t* W1t  = (bf16_t*)(ws + OFF_W1T);
    bf16_t* W2t  = (bf16_t*)(ws + OFF_W2T);
    bf16_t* ctxb = (bf16_t*)(ws + OFF_CTXB);
    float*  out  = (float*)d_out;

    // 1) convert query to bf16
    cvt_f32_bf16<<<4096, 256, 0, stream>>>(query, Xb, (M_ROWS * D) / 8);
    // 2) transpose+convert weights: qkv_w [D x 3D] -> W1t [3D x D]; out_w [D x D] -> W2t
    transpose_f32_bf16<<<dim3(N_QKV / 32, D / 32), dim3(32, 8), 0, stream>>>(qkv_w, W1t, D, N_QKV);
    transpose_f32_bf16<<<dim3(D / 32, D / 32), dim3(32, 8), 0, stream>>>(out_w, W2t, D, D);
    // 3) GEMM1: qkvf = Xb @ W1 + qkv_b   [8192 x 6144]
    gemm_bf16_wmma<<<dim3(N_QKV / 128, M_ROWS / 128), 256, 0, stream>>>(
        Xb, W1t, qkv_b, qkvf, M_ROWS, N_QKV, D);
    // 4) per-token attention, writes permuted bf16 context
    attn_kernel<<<M_ROWS, 256, 0, stream>>>(qkvf, mask, ctxb);
    // 5) GEMM2: out = ctx @ out_w + out_b   [8192 x 2048]
    gemm_bf16_wmma<<<dim3(D / 128, M_ROWS / 128), 256, 0, stream>>>(
        ctxb, W2t, out_b, out, M_ROWS, D, D);
}
